// MultiHeadAttention_17901423690378
// MI455X (gfx1250) — matrix operand order repaired and checked by reading, not compiled, not run on hardware
//
#include <hip/hip_runtime.h>

#define DEV __device__ __forceinline__

typedef __attribute__((ext_vector_type(16))) __bf16 v16bf;
typedef __attribute__((ext_vector_type(8)))  float  v8f;
typedef __attribute__((ext_vector_type(8)))  int    v8i;
typedef __attribute__((ext_vector_type(4)))  unsigned int u32x4;
typedef __attribute__((ext_vector_type(4)))  int    i32x4;
typedef __attribute__((ext_vector_type(8)))  int    i32x8;

constexpr int BATCH = 4, SEQ = 1024, DIM = 1024, NH = 16, HDIM = 64;
constexpr int LDSW = 17;  // dwords per 32-element K-chunk row (16 bf16-pairs + 1 pad)

DEV unsigned short f2bf(float f) {
  unsigned int u = __builtin_bit_cast(unsigned int, f);
  unsigned int r = (u + 0x7FFFu + ((u >> 16) & 1u)) >> 16;  // round-to-nearest-even
  return (unsigned short)r;
}
DEV unsigned int pack2(float lo, float hi) {
  return (unsigned int)f2bf(lo) | ((unsigned int)f2bf(hi) << 16);
}

DEV unsigned lds_off(const void* p) { return (unsigned)(uintptr_t)p; }

// ---- Tensor Data Mover: 2-D tile load Global->LDS with 1-dword pad per 16 dwords ----
// data elements are bf16 (2 bytes). tile_d0 elements/row, tile_d1 rows,
// stride_el = tensor dim0 stride in elements. LDS row group stride becomes 17 dwords
// per 16-dword chunk (matches our padded fragment layout).
DEV void tdm_load_2d_bf16(unsigned ldsbyte, const void* gptr, unsigned tensor_d0,
                          unsigned tensor_d1, unsigned tile_d0, unsigned tile_d1,
                          unsigned stride_el) {
  unsigned long long ga = (unsigned long long)(uintptr_t)gptr;
  u32x4 g0;
  g0[0] = 1u;                                   // count=1, user mode, no gather
  g0[1] = ldsbyte;                              // lds_addr
  g0[2] = (unsigned)(ga & 0xFFFFFFFFu);         // global_addr[31:0]
  g0[3] = (unsigned)((ga >> 32) & 0x1FFFFFFu) | (2u << 30);  // addr[56:32] | type=2
  i32x8 g1;
  g1[0] = (int)((1u << 16) | (1u << 20) | (3u << 22));  // data_size=2B, pad_en, ivl=16dw (amount=1dw)
  g1[1] = (int)((tensor_d0 & 0xFFFFu) << 16);                        // tensor_dim0 lo
  g1[2] = (int)(((tensor_d0 >> 16) & 0xFFFFu) | ((tensor_d1 & 0xFFFFu) << 16));
  g1[3] = (int)(((tensor_d1 >> 16) & 0xFFFFu) | ((tile_d0 & 0xFFFFu) << 16));
  g1[4] = (int)(tile_d1 & 0xFFFFu);             // tile_dim1; tile_dim2=0 (2-D)
  g1[5] = (int)stride_el;                       // tensor_dim0_stride lo
  g1[6] = 0;                                    // stride hi / dim1_stride lo
  g1[7] = 0;
  i32x4 z4 = {0, 0, 0, 0};
#if defined(__clang_major__) && __clang_major__ >= 23
  i32x8 z8 = {0, 0, 0, 0, 0, 0, 0, 0};
  __builtin_amdgcn_tensor_load_to_lds(g0, g1, z4, z4, z8, 0);
#else
  __builtin_amdgcn_tensor_load_to_lds(g0, g1, z4, z4, 0);
#endif
}

// A fragment (16x32 bf16): lane holds row = base + lane%16; g = lane/16.
// VGPR v holds K pair index (v/4)*8 + (v%4) + 4g.  stride/koff in dwords.
DEV v16bf frag_a(const unsigned int* lds, int rowbase, int lane, int stride, int koff) {
  const unsigned int* r = lds + (rowbase + (lane & 15)) * stride + koff;
  int g4 = (lane >> 4) << 2;
  v8i p;
#pragma unroll
  for (int v = 0; v < 8; ++v)
    p[v] = (int)r[((v >> 2) << 3) + (v & 3) + g4];
  return __builtin_bit_cast(v16bf, p);
}
// B fragment (32x16 bf16): lane holds col n = base + lane%16; VGPR v holds K pair (v/4)*8 + (v%4) + 4g.
DEV v16bf frag_b(const unsigned int* lds, int rowbase, int lane, int stride, int koff) {
  const unsigned int* r = lds + (rowbase + (lane & 15)) * stride + koff;
  int g4 = (lane >> 4) << 2;
  v8i p;
#pragma unroll
  for (int v = 0; v < 8; ++v)
    p[v] = (int)r[((v >> 2) << 3) + (v & 3) + g4];
  return __builtin_bit_cast(v16bf, p);
}

DEV v8f wmma_bf16(v16bf a, v16bf b, v8f c) {
  return __builtin_amdgcn_wmma_f32_16x16x32_bf16(false, a, false, b, (short)0, c,
                                                 false, false);
}

// Stage ROWS x 32 tile (row-major, fp32) into LDS as bf16 pairs, padded rows.
template <int ROWS>
DEV void stage_f32(const float* __restrict__ src, int ld, unsigned int* lds, int t) {
#pragma unroll
  for (int i = 0; i < (ROWS * 16) / 256; ++i) {
    int p = i * 256 + t;
    int row = p >> 4, kp = p & 15;
    const float* s = src + (size_t)row * ld + kp * 2;
    lds[row * LDSW + kp] = pack2(s[0], s[1]);
  }
}
// B source stored K-major (32 x 64, e.g. V): transpose into LDS [n][k].
DEV void stage_b_kmajor_bf16(const unsigned short* __restrict__ src, int ld,
                             unsigned int* lds, int t) {
  unsigned short* l16 = (unsigned short*)lds;
#pragma unroll
  for (int i = 0; i < 8; ++i) {
    int e = i * 256 + t;
    int n = e & 63, k = e >> 6;
    l16[n * (2 * LDSW) + k] = src[(size_t)k * ld + n];
  }
}

// ---------------- Kernel 1: fused Q/K/V projections ----------------
__global__ __launch_bounds__(256) void proj_qkv_kernel(
    const float* __restrict__ q_in, const float* __restrict__ k_in,
    const float* __restrict__ v_in, const float* __restrict__ Wq,
    const float* __restrict__ Wk, const float* __restrict__ Wv,
    const float* __restrict__ bq, const float* __restrict__ bk,
    const float* __restrict__ bv, unsigned short* __restrict__ ws) {
  __shared__ unsigned int ldsA[128 * LDSW];
  __shared__ unsigned int ldsB[128 * LDSW];
  int z = blockIdx.z;
  const float* X = (z == 0) ? q_in : (z == 1) ? k_in : v_in;
  const float* W = (z == 0) ? Wq : (z == 1) ? Wk : Wv;
  const float* bias = (z == 0) ? bq : (z == 1) ? bk : bv;
  unsigned short* out = ws + (size_t)z * (size_t)(BATCH * SEQ * DIM);

  int t = threadIdx.x, lane = t & 31, wid = t >> 5;
  int wm = (wid & 3) * 32, wn = (wid >> 2) * 64;
  int m0 = blockIdx.x * 128, n0 = blockIdx.y * 128;

  v8f acc[2][4] = {};
  for (int kk = 0; kk < DIM; kk += 32) {
    stage_f32<128>(X + (size_t)m0 * DIM + kk, DIM, ldsA, t);
    stage_f32<128>(W + (size_t)n0 * DIM + kk, DIM, ldsB, t);
    __syncthreads();
    v16bf a[2], b[4];
#pragma unroll
    for (int i = 0; i < 2; ++i) a[i] = frag_a(ldsA, wm + i * 16, lane, LDSW, 0);
#pragma unroll
    for (int j = 0; j < 4; ++j) b[j] = frag_b(ldsB, wn + j * 16, lane, LDSW, 0);
#pragma unroll
    for (int i = 0; i < 2; ++i)
#pragma unroll
      for (int j = 0; j < 4; ++j) acc[i][j] = wmma_bf16(a[i], b[j], acc[i][j]);
    __syncthreads();
  }

  int g = lane >> 4, nc = lane & 15;
#pragma unroll
  for (int i = 0; i < 2; ++i)
#pragma unroll
    for (int j = 0; j < 4; ++j) {
      int ncol = n0 + wn + j * 16 + nc;
      int h = ncol >> 6, hd = ncol & 63;
      float bb = bias[ncol];
#pragma unroll
      for (int r = 0; r < 8; ++r) {
        int mrow = m0 + wm + i * 16 + r + 8 * g;
        int bidx = mrow >> 10, s = mrow & 1023;
        size_t oidx = ((size_t)((bidx * NH + h) * SEQ + s)) * HDIM + hd;
        out[oidx] = f2bf(acc[i][j][r] + bb);
      }
    }
}

// ---------------- Kernel 2: scores = mask(scale * Q K^T), TDM-staged ----------------
__global__ __launch_bounds__(256) void scores_kernel(
    const unsigned short* __restrict__ Qws, const unsigned short* __restrict__ Kws,
    const int* __restrict__ mask, float* __restrict__ attnW) {
  // Whole 128x64 panels, padded row stride = 2*17 dwords.
  __shared__ unsigned int ldsA[128 * 2 * LDSW];
  __shared__ unsigned int ldsB[128 * 2 * LDSW];
  int z = blockIdx.z;  // b*H + h
  int b = z >> 4;
  const unsigned short* Q = Qws + (size_t)z * SEQ * HDIM;
  const unsigned short* K = Kws + (size_t)z * SEQ * HDIM;
  int t = threadIdx.x, lane = t & 31, wid = t >> 5;
  int wm = (wid & 3) * 32, wn = (wid >> 2) * 64;
  int m0 = blockIdx.x * 128, n0 = blockIdx.y * 128;

  if (wid == 0) {  // one wave issues both TDM panel loads
    tdm_load_2d_bf16(lds_off(ldsA), Q + (size_t)m0 * HDIM, HDIM, SEQ - m0, HDIM, 128,
                     HDIM);
    tdm_load_2d_bf16(lds_off(ldsB), K + (size_t)n0 * HDIM, HDIM, SEQ - n0, HDIM, 128,
                     HDIM);
    __builtin_amdgcn_s_wait_tensorcnt(0);
  }
  __syncthreads();

  v8f acc[2][4] = {};
#pragma unroll
  for (int ks = 0; ks < 2; ++ks) {  // K = 64 = 2 chunks of 32, all in LDS already
    int koff = ks * LDSW;
    v16bf a[2], bb[4];
#pragma unroll
    for (int i = 0; i < 2; ++i) a[i] = frag_a(ldsA, wm + i * 16, lane, 2 * LDSW, koff);
#pragma unroll
    for (int j = 0; j < 4; ++j) bb[j] = frag_b(ldsB, wn + j * 16, lane, 2 * LDSW, koff);
#pragma unroll
    for (int i = 0; i < 2; ++i)
#pragma unroll
      for (int j = 0; j < 4; ++j) acc[i][j] = wmma_bf16(a[i], bb[j], acc[i][j]);
  }

  const float scale = 0.125f;  // 64^-0.5
  int g = lane >> 4, nc = lane & 15;
#pragma unroll
  for (int i = 0; i < 2; ++i)
#pragma unroll
    for (int j = 0; j < 4; ++j) {
      int kcol = n0 + wn + j * 16 + nc;
#pragma unroll
      for (int r = 0; r < 8; ++r) {
        int qrow = m0 + wm + i * 16 + r + 8 * g;
        int mv = mask[((size_t)b << 20) + ((size_t)qrow << 10) + kcol];
        float vsc = (mv == 0) ? -1e9f : acc[i][j][r] * scale;
        attnW[((size_t)z << 20) + ((size_t)qrow << 10) + kcol] = vsc;
      }
    }
}

// ---------------- Kernel 3: row softmax (one wave per 1024-row) ----------------
__global__ __launch_bounds__(256) void softmax_kernel(float* __restrict__ attnW) {
  int lane = threadIdx.x & 31, wid = threadIdx.x >> 5;
  size_t row = (size_t)blockIdx.x * 8 + wid;
  float* p = attnW + row * SEQ;
  float4 x[8];
  float mx = -3.4e38f;
#pragma unroll
  for (int i = 0; i < 8; ++i) {
    x[i] = ((const float4*)p)[i * 32 + lane];
    mx = fmaxf(mx, fmaxf(fmaxf(x[i].x, x[i].y), fmaxf(x[i].z, x[i].w)));
  }
#pragma unroll
  for (int o = 16; o > 0; o >>= 1) mx = fmaxf(mx, __shfl_xor(mx, o, 32));
  float sum = 0.f;
#pragma unroll
  for (int i = 0; i < 8; ++i) {
    x[i].x = __expf(x[i].x - mx);
    x[i].y = __expf(x[i].y - mx);
    x[i].z = __expf(x[i].z - mx);
    x[i].w = __expf(x[i].w - mx);
    sum += x[i].x + x[i].y + x[i].z + x[i].w;
  }
#pragma unroll
  for (int o = 16; o > 0; o >>= 1) sum += __shfl_xor(sum, o, 32);
  float inv = 1.0f / sum;
#pragma unroll
  for (int i = 0; i < 8; ++i) {
    x[i].x *= inv; x[i].y *= inv; x[i].z *= inv; x[i].w *= inv;
    ((float4*)p)[i * 32 + lane] = x[i];
  }
}

// ---------------- Kernel 4: attn_out = P @ V ----------------
__global__ __launch_bounds__(256) void pv_kernel(const float* __restrict__ attnW,
                                                 const unsigned short* __restrict__ Vws,
                                                 unsigned short* __restrict__ AO) {
  __shared__ unsigned int ldsA[128 * LDSW];
  __shared__ unsigned int ldsB[64 * LDSW];
  int z = blockIdx.z, b = z >> 4, h = z & 15;
  const float* P = attnW + ((size_t)z << 20);
  const unsigned short* V = Vws + (size_t)z * SEQ * HDIM;
  int t = threadIdx.x, lane = t & 31, wid = t >> 5;
  int wm = (wid & 3) * 32, wn = (wid >> 2) * 32;
  int m0 = blockIdx.x * 128;

  v8f acc[2][2] = {};
  for (int kk = 0; kk < SEQ; kk += 32) {
    stage_f32<128>(P + (size_t)m0 * SEQ + kk, SEQ, ldsA, t);
    stage_b_kmajor_bf16(V + (size_t)kk * HDIM, HDIM, ldsB, t);
    __syncthreads();
    v16bf a[2], bb[2];
#pragma unroll
    for (int i = 0; i < 2; ++i) a[i] = frag_a(ldsA, wm + i * 16, lane, LDSW, 0);
#pragma unroll
    for (int j = 0; j < 2; ++j) bb[j] = frag_b(ldsB, wn + j * 16, lane, LDSW, 0);
#pragma unroll
    for (int i = 0; i < 2; ++i)
#pragma unroll
      for (int j = 0; j < 2; ++j) acc[i][j] = wmma_bf16(a[i], bb[j], acc[i][j]);
    __syncthreads();
  }

  int g = lane >> 4, nc = lane & 15;
#pragma unroll
  for (int i = 0; i < 2; ++i)
#pragma unroll
    for (int j = 0; j < 2; ++j) {
      int hd = wn + j * 16 + nc;
#pragma unroll
      for (int r = 0; r < 8; ++r) {
        int qrow = m0 + wm + i * 16 + r + 8 * g;
        AO[((size_t)(b * SEQ + qrow)) * DIM + h * HDIM + hd] = f2bf(acc[i][j][r]);
      }
    }
}

// ---------------- Kernel 5: output = attn_out @ Wo^T + bo (TDM-staged A) ----------------
__global__ __launch_bounds__(256) void outproj_kernel(
    const unsigned short* __restrict__ AO, const float* __restrict__ Wo,
    const float* __restrict__ bo, float* __restrict__ out) {
  __shared__ unsigned int ldsA[128 * LDSW];
  __shared__ unsigned int ldsB[128 * LDSW];
  int t = threadIdx.x, lane = t & 31, wid = t >> 5;
  int wm = (wid & 3) * 32, wn = (wid >> 2) * 64;
  int m0 = blockIdx.x * 128, n0 = blockIdx.y * 128;

  v8f acc[2][4] = {};
  for (int kk = 0; kk < DIM; kk += 32) {
    stage_f32<128>(Wo + (size_t)n0 * DIM + kk, DIM, ldsB, t);
    if (wid == 0) {  // TDM: 128x32 bf16 tile, row stride DIM elements
      tdm_load_2d_bf16(lds_off(ldsA), AO + (size_t)m0 * DIM + kk, DIM - kk,
                       BATCH * SEQ - m0, 32, 128, DIM);
      __builtin_amdgcn_s_wait_tensorcnt(0);
    }
    __syncthreads();
    v16bf a[2], bb[4];
#pragma unroll
    for (int i = 0; i < 2; ++i) a[i] = frag_a(ldsA, wm + i * 16, lane, LDSW, 0);
#pragma unroll
    for (int j = 0; j < 4; ++j) bb[j] = frag_b(ldsB, wn + j * 16, lane, LDSW, 0);
#pragma unroll
    for (int i = 0; i < 2; ++i)
#pragma unroll
      for (int j = 0; j < 4; ++j) acc[i][j] = wmma_bf16(a[i], bb[j], acc[i][j]);
    __syncthreads();
  }

  int g = lane >> 4, nc = lane & 15;
#pragma unroll
  for (int i = 0; i < 2; ++i)
#pragma unroll
    for (int j = 0; j < 4; ++j) {
      int ncol = n0 + wn + j * 16 + nc;
      float bb2 = bo[ncol];
#pragma unroll
      for (int r = 0; r < 8; ++r) {
        int mrow = m0 + wm + i * 16 + r + 8 * g;
        out[(size_t)mrow * DIM + ncol] = acc[i][j][r] + bb2;
      }
    }
}

extern "C" void kernel_launch(void* const* d_in, const int* in_sizes, int n_in,
                              void* d_out, int out_size, void* d_ws, size_t ws_size,
                              hipStream_t stream) {
  const float* query  = (const float*)d_in[0];
  const float* key_in = (const float*)d_in[1];
  const float* value  = (const float*)d_in[2];
  const int*   mask   = (const int*)d_in[3];
  const float* Wq = (const float*)d_in[4];
  const float* bq = (const float*)d_in[5];
  const float* Wk = (const float*)d_in[6];
  const float* bk = (const float*)d_in[7];
  const float* Wv = (const float*)d_in[8];
  const float* bv = (const float*)d_in[9];
  const float* Wo = (const float*)d_in[10];
  const float* bo = (const float*)d_in[11];

  float* out   = (float*)d_out;
  float* attnW = out + (size_t)BATCH * SEQ * DIM;  // attn_weights region (B,H,S,S)

  unsigned short* Qws  = (unsigned short*)d_ws;                 // (B,H,S,64) bf16
  unsigned short* Kws  = Qws + (size_t)BATCH * SEQ * DIM;
  unsigned short* Vws  = Kws + (size_t)BATCH * SEQ * DIM;
  unsigned short* AOws = Vws + (size_t)BATCH * SEQ * DIM;       // (B,S,D) bf16

  dim3 blk(256);
  proj_qkv_kernel<<<dim3(BATCH * SEQ / 128, DIM / 128, 3), blk, 0, stream>>>(
      query, key_in, value, Wq, Wk, Wv, bq, bk, bv, Qws);
  scores_kernel<<<dim3(SEQ / 128, SEQ / 128, BATCH * NH), blk, 0, stream>>>(
      Qws, Kws, mask, attnW);
  softmax_kernel<<<dim3(BATCH * NH * SEQ / 8), blk, 0, stream>>>(attnW);
  pv_kernel<<<dim3(SEQ / 128, 1, BATCH * NH), blk, 0, stream>>>(attnW, Vws, AOws);
  outproj_kernel<<<dim3(BATCH * SEQ / 128, DIM / 128, 1), blk, 0, stream>>>(
      AOws, Wo, bo, out);
}
